// KernelRegressorAttention_55825984914048
// MI455X (gfx1250) — matrix operand order repaired and checked by reading, not compiled, not run on hardware
//
#include <hip/hip_runtime.h>
#include <hip/hip_bf16.h>

// ---- CDNA5 WMMA types ----
typedef __attribute__((ext_vector_type(16))) __bf16 v16bf;
typedef __attribute__((ext_vector_type(8)))  __bf16 v8bf;
typedef __attribute__((ext_vector_type(8)))  float  v8f;

#define NHEADS 32          // B*H = 2*16
#define SEQ    4096
#define DIM    128         // head dim D
#define FEAT   256         // feature dim F
#define HIDW   256         // hidden width
#define TTOK   64          // tokens per phi block
#define EPSV   1e-6f

// workspace layout (bytes); total ~141 MB
#define OFF_W1P    0u          // bf16 packed W1 fragments: 64 KB
#define OFF_W2P    65536u      // bf16 packed W2 fragments: 128 KB
#define OFF_KV     196608u     // fp32 kv: 32*256*128 = 4 MB (plain stores)
#define OFF_KSUM   4390912u    // fp32 ksum: 32 KB
#define OFF_KVP    4423680u    // bf16 packed kv fragments: 2 MB
#define OFF_PHIQ   6520832u    // bf16 phi(q) [head][N][F]: 64 MB
#define OFF_PHIKT  73629696u   // bf16 phi(k)^T [head][F][N]: 64 MB

__device__ __forceinline__ v8f wmma_bf16(v16bf a, v16bf b, v8f c) {
  return __builtin_amdgcn_wmma_f32_16x16x32_bf16(false, a, false, b, (short)0, c,
                                                 false, false);
}

__device__ __forceinline__ v16bf cat8(v8bf lo, v8bf hi) {
  return __builtin_shufflevector(lo, hi, 0, 1, 2, 3, 4, 5, 6, 7,
                                         8, 9, 10, 11, 12, 13, 14, 15);
}

// phi for a 16-token tile, fully via WMMA (identical to the round-1 version
// that compiled without spills).  Output p[16] tiles in C/D layout:
// lane = column (nt*16 + lane%16), VGPR r = token row r + 8*(lane/16).
__device__ __forceinline__ void phi16(const float* __restrict__ x,
                                      const __bf16* __restrict__ W1p,
                                      const __bf16* __restrict__ W2p,
                                      const float* __restrict__ b1,
                                      const float* __restrict__ b2,
                                      __bf16* stage, int lane, v8f p[16]) {
  const int m = lane & 15;
  const int g = lane >> 4;

  // A-fragments of X (4 fragments over D=128)
  v16bf aX[4];
#pragma unroll
  for (int kt = 0; kt < 4; ++kt) {
#pragma unroll
    for (int e = 0; e < 16; ++e) {
      int kk = kt * 32 + (e < 8 ? (e + 8 * g) : (16 + (e - 8) + 8 * g));
      aX[kt][e] = (__bf16)x[m * DIM + kk];
    }
  }

  // H = silu(X @ W1 + b1)
#pragma unroll
  for (int nt = 0; nt < 16; ++nt) {
    v8f acc = {0.f, 0.f, 0.f, 0.f, 0.f, 0.f, 0.f, 0.f};
#pragma unroll
    for (int kt = 0; kt < 4; ++kt) {
      v16bf b = *(const v16bf*)(W1p + (size_t)(nt * 4 + kt) * 512 + lane * 16);
      acc = wmma_bf16(aX[kt], b, acc);
    }
    float bias = b1[nt * 16 + m];
#pragma unroll
    for (int r = 0; r < 8; ++r) {
      float f = acc[r] + bias;
      float h = f / (1.0f + __expf(-f));
      stage[(r + 8 * g) * HIDW + nt * 16 + m] = (__bf16)h;
    }
  }
  __syncthreads();

  v16bf aH[8];
#pragma unroll
  for (int kt = 0; kt < 8; ++kt) {
    v8bf lo = *(const v8bf*)(stage + m * HIDW + kt * 32 + 8 * g);
    v8bf hi = *(const v8bf*)(stage + m * HIDW + kt * 32 + 16 + 8 * g);
    aH[kt] = cat8(lo, hi);
  }
  __syncthreads();  // stage regions reusable by caller after this

  // P = elu(H @ W2 + b2) + 1
#pragma unroll
  for (int nt = 0; nt < 16; ++nt) {
    v8f acc = {0.f, 0.f, 0.f, 0.f, 0.f, 0.f, 0.f, 0.f};
#pragma unroll
    for (int kt = 0; kt < 8; ++kt) {
      v16bf b = *(const v16bf*)(W2p + (size_t)(nt * 8 + kt) * 512 + lane * 16);
      acc = wmma_bf16(aH[kt], b, acc);
    }
    float bias = b2[nt * 16 + m];
#pragma unroll
    for (int r = 0; r < 8; ++r) {
      float y = acc[r] + bias;
      acc[r] = (y > 0.0f) ? (y + 1.0f) : __expf(y);
    }
    p[nt] = acc;
  }

  // L2 normalize per token row
  float ss[8];
#pragma unroll
  for (int r = 0; r < 8; ++r) ss[r] = 0.0f;
#pragma unroll
  for (int nt = 0; nt < 16; ++nt)
#pragma unroll
    for (int r = 0; r < 8; ++r) ss[r] += p[nt][r] * p[nt][r];
#pragma unroll
  for (int r = 0; r < 8; ++r) {
    float s = ss[r];
    s += __shfl_xor(s, 1, 32);
    s += __shfl_xor(s, 2, 32);
    s += __shfl_xor(s, 4, 32);
    s += __shfl_xor(s, 8, 32);
    ss[r] = 1.0f / (sqrtf(s) + EPSV);
  }
#pragma unroll
  for (int nt = 0; nt < 16; ++nt)
#pragma unroll
    for (int r = 0; r < 8; ++r) p[nt][r] *= ss[r];
}

// --------- phi kernel: MODE 0 -> phi(q) row-major; MODE 1 -> phi(k)^T + ksum
template <int MODE>
__global__ __launch_bounds__(128) void phi_kernel(
    const float* __restrict__ xin, const __bf16* __restrict__ W1p,
    const __bf16* __restrict__ W2p, const float* __restrict__ b1,
    const float* __restrict__ b2, __bf16* __restrict__ phiq,
    __bf16* __restrict__ phikt, float* __restrict__ ksum) {
  __shared__ __align__(64) __bf16 smem[16384];  // 4 wave stages / phiT alias
  const int tid = threadIdx.x, wave = tid >> 5, lane = tid & 31;
  const int m = lane & 15, g = lane >> 4;
  const int head = blockIdx.y, chunk = blockIdx.x;
  __bf16* stage = smem + wave * 4096;

  const float* x =
      xin + (size_t)head * SEQ * DIM + (size_t)(chunk * TTOK + wave * 16) * DIM;
  v8f p[16];
  phi16(x, W1p, W2p, b1, b2, stage, lane, p);

  if (MODE == 0) {
    // write p row-major [16 x 256] into this wave's stage, then copy out
#pragma unroll
    for (int nt = 0; nt < 16; ++nt)
#pragma unroll
      for (int r = 0; r < 8; ++r)
        stage[(r + 8 * g) * FEAT + nt * 16 + m] = (__bf16)p[nt][r];
    // coalesced copy: 16 rows x 512B contiguous at destination
    __bf16* dst = phiq + (size_t)head * SEQ * FEAT +
                  (size_t)(chunk * TTOK + wave * 16) * FEAT;
    for (int i = lane; i < (16 * FEAT) / 8; i += 32)
      ((v8bf*)dst)[i] = ((const v8bf*)stage)[i];
  } else {
    // ksum: column sums over this 16-token tile (few atomics, negligible)
    float* ks = ksum + (size_t)head * FEAT;
#pragma unroll
    for (int nt = 0; nt < 16; ++nt) {
      float s = 0.0f;
#pragma unroll
      for (int r = 0; r < 8; ++r) s += p[nt][r];
      s += __shfl_xor(s, 16, 32);
      if (g == 0) atomicAdd(&ks[nt * 16 + m], s);
    }
    // phiT in LDS: [256][64] bf16 (all waves past phi16's trailing barrier)
    __bf16* phiT = smem;
#pragma unroll
    for (int nt = 0; nt < 16; ++nt)
#pragma unroll
      for (int r = 0; r < 8; ++r)
        phiT[(nt * 16 + m) * TTOK + wave * 16 + r + 8 * g] = (__bf16)p[nt][r];
    __syncthreads();
    // copy out: row f -> phikt[head][f][chunk*64 .. +64], 128B per row segment
    __bf16* base = phikt + (size_t)head * FEAT * SEQ + (size_t)(chunk * TTOK);
    for (int f = tid; f < FEAT; f += 128) {
      const v8bf* src = (const v8bf*)(phiT + f * TTOK);
      v8bf* dst = (v8bf*)(base + (size_t)f * SEQ);
#pragma unroll
      for (int j = 0; j < 8; ++j) dst[j] = src[j];
    }
  }
}

// --------- kv GEMM: kv[F][Dv] = phi(k)^T[F][N] @ v[N][Dv], K=4096 streamed
// grid (4 F-slices, 32 heads, 2 Dv-halves), 128 threads.
// wave w: F rows [slice*64 + w*16, +16); 4 Dv tiles -> 32 persistent VGPRs.
__global__ __launch_bounds__(128) void kv_kernel(const __bf16* __restrict__ phikt,
                                                 const float* __restrict__ vin,
                                                 float* __restrict__ kv) {
  __shared__ __align__(64) __bf16 vT[64 * 32];  // [64 Dv][32 tok] = 4 KB
  const int tid = threadIdx.x, wave = tid >> 5, lane = tid & 31;
  const int m = lane & 15, g = lane >> 4;
  const int head = blockIdx.y, fslice = blockIdx.x, dvhalf = blockIdx.z;
  const int frow = fslice * 64 + wave * 16;  // this wave's F-row base
  const int dv0 = dvhalf * 64;

  v8f acc[4];
#pragma unroll
  for (int i = 0; i < 4; ++i)
    acc[i] = (v8f){0.f, 0.f, 0.f, 0.f, 0.f, 0.f, 0.f, 0.f};

  const __bf16* arow =
      phikt + (size_t)head * FEAT * SEQ + (size_t)(frow + m) * SEQ;
  const float* vh = vin + (size_t)head * SEQ * DIM;

  for (int k0 = 0; k0 < SEQ; k0 += 32) {
    __syncthreads();  // previous iteration's vT reads complete
    // stage v^T chunk: vT[dv_local][t] over 32 tokens x 64 Dv columns
    for (int i = tid; i < 64 * 32; i += 128) {
      int dv = i >> 5, t = i & 31;
      vT[i] = (__bf16)vh[(size_t)(k0 + t) * DIM + dv0 + dv];
    }
    __syncthreads();

    // A fragment straight from global phi(k)^T (two b128 loads)
    v8bf lo = *(const v8bf*)(arow + k0 + 8 * g);
    v8bf hi = *(const v8bf*)(arow + k0 + 16 + 8 * g);
    v16bf a = cat8(lo, hi);
#pragma unroll
    for (int nt = 0; nt < 4; ++nt) {
      v16bf b = cat8(*(const v8bf*)(vT + (nt * 16 + m) * 32 + 8 * g),
                     *(const v8bf*)(vT + (nt * 16 + m) * 32 + 16 + 8 * g));
      acc[nt] = wmma_bf16(a, b, acc[nt]);
    }
  }

  // plain coalesced fp32 stores (no atomics)
  float* dst = kv + (size_t)head * FEAT * DIM;
#pragma unroll
  for (int nt = 0; nt < 4; ++nt)
#pragma unroll
    for (int r = 0; r < 8; ++r)
      dst[(size_t)(frow + r + 8 * g) * DIM + dv0 + nt * 16 + m] = acc[nt][r];
}

// --------- output: out = (phi_q @ kv) / max(phi_q . ksum, eps)
__global__ __launch_bounds__(128) void out_kernel(
    const __bf16* __restrict__ phiq, const __bf16* __restrict__ kvp,
    const float* __restrict__ ksum, float* __restrict__ out) {
  __shared__ float ksumL[FEAT];
  __shared__ float denL[4][16];
  const int tid = threadIdx.x, wave = tid >> 5, lane = tid & 31;
  const int m = lane & 15, g = lane >> 4;
  const int head = blockIdx.y, chunk = blockIdx.x;

  for (int i = tid; i < FEAT; i += 128) ksumL[i] = ksum[(size_t)head * FEAT + i];
  __syncthreads();

  // A-fragments of phi_q from global; fuse den partial (this lane's K-slice)
  const __bf16* prow = phiq + (size_t)head * SEQ * FEAT +
                       (size_t)(chunk * TTOK + wave * 16 + m) * FEAT;
  v16bf aP[8];
  float dpart = 0.0f;
#pragma unroll
  for (int kt = 0; kt < 8; ++kt) {
    v8bf lo = *(const v8bf*)(prow + kt * 32 + 8 * g);
    v8bf hi = *(const v8bf*)(prow + kt * 32 + 16 + 8 * g);
    aP[kt] = cat8(lo, hi);
#pragma unroll
    for (int e = 0; e < 8; ++e) {
      dpart += (float)lo[e] * ksumL[kt * 32 + 8 * g + e];
      dpart += (float)hi[e] * ksumL[kt * 32 + 16 + 8 * g + e];
    }
  }
  dpart += __shfl_xor(dpart, 16, 32);  // combine both K-halves -> den of row m
  if (g == 0) denL[wave][m] = dpart;
  __syncthreads();
  float deninv[8];
#pragma unroll
  for (int r = 0; r < 8; ++r)
    deninv[r] = 1.0f / fmaxf(denL[wave][r + 8 * g], EPSV);

  // num = phi_q(16x256) @ kv(256x128): 8 N-tiles x 8 K-steps
  const __bf16* kvh = kvp + (size_t)head * FEAT * DIM;
  float* oB =
      out + (size_t)head * SEQ * DIM + (size_t)(chunk * TTOK + wave * 16) * DIM;
#pragma unroll
  for (int nt = 0; nt < 8; ++nt) {
    v8f acc = {0.f, 0.f, 0.f, 0.f, 0.f, 0.f, 0.f, 0.f};
#pragma unroll
    for (int kt = 0; kt < 8; ++kt) {
      v16bf b = *(const v16bf*)(kvh + (size_t)(nt * 8 + kt) * 512 + lane * 16);
      acc = wmma_bf16(aP[kt], b, acc);
    }
#pragma unroll
    for (int r = 0; r < 8; ++r)
      oB[(r + 8 * g) * DIM + nt * 16 + m] = acc[r] * deninv[r];
  }
}

// ---------------- helpers ----------------
__global__ void zero_kernel(float* p, int n) {
  int i = blockIdx.x * 256 + threadIdx.x;
  if (i < n) p[i] = 0.0f;
}

__global__ void packw_kernel(const float* __restrict__ W1,
                             const float* __restrict__ W2,
                             __bf16* __restrict__ W1p,
                             __bf16* __restrict__ W2p) {
  int tid = blockIdx.x * 256 + threadIdx.x;
  const int n1 = 128 * 256;
  if (tid < n1) {
    int o = tid, frag = o >> 9, lane = (o >> 4) & 31, e = o & 15;
    int kt = frag & 3, nt = frag >> 2, g = lane >> 4;
    int n = nt * 16 + (lane & 15), kk = kt * 32 + 8 * g + e + (e >= 8 ? 8 : 0);
    W1p[o] = (__bf16)W1[kk * 256 + n];
  } else if (tid < n1 + 256 * 256) {
    int o = tid - n1, frag = o >> 9, lane = (o >> 4) & 31, e = o & 15;
    int kt = frag & 7, nt = frag >> 3, g = lane >> 4;
    int n = nt * 16 + (lane & 15), kk = kt * 32 + 8 * g + e + (e >= 8 ? 8 : 0);
    W2p[o] = (__bf16)W2[kk * 256 + n];
  }
}

__global__ void packkv_kernel(const float* __restrict__ kv,
                              __bf16* __restrict__ kvp) {
  int idx = blockIdx.x * 256 + threadIdx.x;  // 32 * 32768 total
  int head = idx >> 15, o = idx & 32767;
  int frag = o >> 9, lane = (o >> 4) & 31, e = o & 15;
  int kt = frag & 7, nt = frag >> 3, g = lane >> 4;
  int Fi = kt * 32 + 8 * g + e + (e >= 8 ? 8 : 0), Dv = nt * 16 + (lane & 15);
  kvp[idx] = (__bf16)kv[(size_t)head * 32768 + Fi * 128 + Dv];
}

extern "C" void kernel_launch(void* const* d_in, const int* in_sizes, int n_in,
                              void* d_out, int out_size, void* d_ws,
                              size_t ws_size, hipStream_t stream) {
  (void)in_sizes; (void)n_in; (void)out_size; (void)ws_size;
  const float* q = (const float*)d_in[0];
  const float* k = (const float*)d_in[1];
  const float* v = (const float*)d_in[2];
  const float* W1 = (const float*)d_in[3];
  const float* b1 = (const float*)d_in[4];
  const float* W2 = (const float*)d_in[5];
  const float* b2 = (const float*)d_in[6];
  float* out = (float*)d_out;

  char* ws = (char*)d_ws;
  __bf16* W1p = (__bf16*)(ws + OFF_W1P);
  __bf16* W2p = (__bf16*)(ws + OFF_W2P);
  float* kv = (float*)(ws + OFF_KV);
  float* ksum = (float*)(ws + OFF_KSUM);
  __bf16* kvp = (__bf16*)(ws + OFF_KVP);
  __bf16* phiq = (__bf16*)(ws + OFF_PHIQ);
  __bf16* phikt = (__bf16*)(ws + OFF_PHIKT);

  // ksum accumulated via atomics -> zero it each call
  zero_kernel<<<(NHEADS * FEAT + 255) / 256, 256, 0, stream>>>(ksum,
                                                               NHEADS * FEAT);
  packw_kernel<<<(128 * 256 + 256 * 256 + 255) / 256, 256, 0, stream>>>(
      W1, W2, W1p, W2p);

  dim3 gphi(SEQ / TTOK, NHEADS);  // 64 x 32 blocks, 4 waves each
  phi_kernel<0><<<gphi, 128, 0, stream>>>(q, W1p, W2p, b1, b2, phiq, phikt,
                                          ksum);
  phi_kernel<1><<<gphi, 128, 0, stream>>>(k, W1p, W2p, b1, b2, phiq, phikt,
                                          ksum);

  dim3 gkv(FEAT / 64, NHEADS, 2);  // 4 x 32 x 2 = 256 blocks
  kv_kernel<<<gkv, 128, 0, stream>>>(phikt, v, kv);

  packkv_kernel<<<(NHEADS * FEAT * DIM) / 256, 256, 0, stream>>>(kv, kvp);

  out_kernel<<<gphi, 128, 0, stream>>>(phiq, kvp, ksum, out);
}
